// PraxisAttention_69818988363932
// MI455X (gfx1250) — matrix operand order repaired and checked by reading, not compiled, not run on hardware
//
#include <hip/hip_runtime.h>
#include <math.h>

typedef __attribute__((ext_vector_type(16))) __bf16 v16bf;
typedef __attribute__((ext_vector_type(8)))  __bf16 v8bf;
typedef __attribute__((ext_vector_type(4)))  __bf16 v4bf;
typedef __attribute__((ext_vector_type(8)))  float  v8f;

#define S_LEN   2048
#define DHEAD   128
#define NH      8
#define BM      64
#define BN      64
#define KROW    136                 // padded K row in bf16 elems (16B-aligned strides)
#define VROW    72                  // padded V^T row
#define PROW    68                  // padded P row (8B-aligned strides)
#define LDS_K_ELEMS (2*64*KROW)     // 17408
#define LDS_V_ELEMS (128*VROW)      // 9216
#define LDS_P_ELEMS (8*16*PROW)     // 8704
#define SCALING 0.08838834764831845f
#define NEGINF  -1e9f

__device__ __forceinline__ v16bf ld_b16x16(const __bf16* p) {
  v8bf lo = *(const v8bf*)p;
  v8bf hi = *(const v8bf*)(p + 16);
  v16bf r;
#pragma unroll
  for (int i = 0; i < 8; ++i) { r[i] = lo[i]; r[i + 8] = hi[i]; }
  return r;
}

__device__ __forceinline__ v16bf ld_afrag(const __bf16* p0, const __bf16* p1) {
  v4bf a = *(const v4bf*)p0, b = *(const v4bf*)(p0 + 4);
  v4bf c = *(const v4bf*)p1, d = *(const v4bf*)(p1 + 4);
  v16bf r;
#pragma unroll
  for (int i = 0; i < 4; ++i) { r[i] = a[i]; r[4+i] = b[i]; r[8+i] = c[i]; r[12+i] = d[i]; }
  return r;
}

__device__ __forceinline__ void cvt4(v16bf& a, int base, float4 f) {
  a[base + 0] = (__bf16)f.x; a[base + 1] = (__bf16)f.y;
  a[base + 2] = (__bf16)f.z; a[base + 3] = (__bf16)f.w;
}

// -------------------- kernel 1: lambda scalar + zero stats --------------------
__global__ __launch_bounds__(128)
void prep_kernel(const float* __restrict__ lq1, const float* __restrict__ lk1,
                 const float* __restrict__ lq2, const float* __restrict__ lk2,
                 float* __restrict__ ws) {
  __shared__ float r1[128], r2[128];
  int i = threadIdx.x;
  r1[i] = lq1[i] * lk1[i];
  r2[i] = lq2[i] * lk2[i];
  __syncthreads();
#pragma unroll
  for (int off = 64; off > 0; off >>= 1) {
    if (i < off) { r1[i] += r1[i + off]; r2[i] += r2[i + off]; }
    __syncthreads();
  }
  if (i == 0) ws[0] = __expf(r1[0]) - __expf(r2[0]) + 0.8f;
  if (i >= 8 && i < 40) ws[i] = 0.0f;   // zero stats[32] at ws[8..39]
}

// -------------------- kernel 2: differential flash attention --------------------
__global__ __launch_bounds__(256)
void diff_flash(const float* __restrict__ q, const float* __restrict__ k,
                const float* __restrict__ v, const float* __restrict__ ws,
                float* __restrict__ O, float* __restrict__ stats) {
  __shared__ __align__(16) char smem[(LDS_K_ELEMS + LDS_V_ELEMS + LDS_P_ELEMS) * 2];
  __bf16* Kt = (__bf16*)smem;                 // [2][64][KROW]
  __bf16* Vt = Kt + LDS_K_ELEMS;              // V^T: [128][VROW]
  __bf16* Pt = Vt + LDS_V_ELEMS;              // [8 waves][16][PROW]
  float*  Comb = (float*)smem;                // reused after kv loop: [64][132]
  __shared__ float s_sum, s_sq;

  const int tid  = threadIdx.x;
  const int wid  = tid >> 5;
  const int lane = tid & 31;
  const int sub  = wid >> 2;        // differential sub-head (0/1)
  const int mt   = wid & 3;         // 16-row query sub-tile
  const int hi   = lane >> 4;
  const int ln   = lane & 15;
  const int koff = hi << 3;         // B-fragment K offset
  const int fb   = hi << 3;         // A-fragment K offset

  const int qt = blockIdx.x & 31;
  const int bh = blockIdx.x >> 5;
  const int b  = bh >> 3;
  const int h  = bh & 7;
  const int q0 = qt * BM;

  if (tid == 0) { s_sum = 0.0f; s_sq = 0.0f; }

  // Q fragments (bf16 A-layout), head 2h+sub, this wave's 16 rows
  const int hq = 2 * h + sub;
  const float* qp = q + (((size_t)b * 16 + hq) * S_LEN + (q0 + mt * 16 + ln)) * DHEAD;
  v16bf qa[4];
#pragma unroll
  for (int kc = 0; kc < 4; ++kc) {
    const float* p1 = qp + kc * 32 + fb;
    const float* p2 = p1 + 16;
    v16bf a;
    cvt4(a, 0,  *(const float4*)p1);
    cvt4(a, 4,  *(const float4*)(p1 + 4));
    cvt4(a, 8,  *(const float4*)p2);
    cvt4(a, 12, *(const float4*)(p2 + 4));
    qa[kc] = a;
  }

  v8f acc[8];
  const v8f vzero = {0, 0, 0, 0, 0, 0, 0, 0};
#pragma unroll
  for (int i = 0; i < 8; ++i) acc[i] = vzero;
  float m[8], l[8];
#pragma unroll
  for (int r = 0; r < 8; ++r) { m[r] = 0.0f; l[r] = 1.0f; }  // ghostmax implicit 0 logit

  const float* kb0 = k + (((size_t)b * 16 + 2 * h)     * S_LEN) * DHEAD;
  const float* kb1 = k + (((size_t)b * 16 + 2 * h + 1) * S_LEN) * DHEAD;
  const float* vb  = v + (((size_t)b * 8  + h)         * S_LEN) * DHEAD;

  for (int nt = 0; nt <= qt; ++nt) {
    const int n0 = nt * BN;
    // cooperative load + f32->bf16 convert into LDS (K both sub-heads, V transposed)
#pragma unroll
    for (int p = 0; p < 8; ++p) {
      int e = p * 1024 + tid * 4;
      int row = e >> 7, col = e & 127;
      float4 x0 = *(const float4*)(kb0 + (size_t)(n0 + row) * DHEAD + col);
      float4 x1 = *(const float4*)(kb1 + (size_t)(n0 + row) * DHEAD + col);
      float4 xv = *(const float4*)(vb  + (size_t)(n0 + row) * DHEAD + col);
      v4bf pk0 = {(__bf16)x0.x, (__bf16)x0.y, (__bf16)x0.z, (__bf16)x0.w};
      v4bf pk1 = {(__bf16)x1.x, (__bf16)x1.y, (__bf16)x1.z, (__bf16)x1.w};
      *(v4bf*)&Kt[row * KROW + col]        = pk0;
      *(v4bf*)&Kt[(64 + row) * KROW + col] = pk1;
      Vt[(col + 0) * VROW + row] = (__bf16)xv.x;
      Vt[(col + 1) * VROW + row] = (__bf16)xv.y;
      Vt[(col + 2) * VROW + row] = (__bf16)xv.z;
      Vt[(col + 3) * VROW + row] = (__bf16)xv.w;
    }
    if (nt < qt) __builtin_prefetch(kb0 + (size_t)(n0 + BN) * DHEAD + tid * 32, 0, 0);
    __syncthreads();

    // ---- scores: S = Q @ K^T, 16 steps (kc-major), 2-deep pipelined B frags ----
    v8f sc[4];
#pragma unroll
    for (int ns = 0; ns < 4; ++ns) sc[ns] = vzero;
    const __bf16* kk = Kt + sub * (64 * KROW);
    auto ldK = [&](int st) -> v16bf {
      int kc = st >> 2, ns = st & 3;
      return ld_b16x16(kk + (ns * 16 + ln) * KROW + kc * 32 + koff);
    };
    {
      v16bf fr0 = ldK(0), fr1 = ldK(1);
#pragma unroll
      for (int st = 0; st < 16; ++st) {
        v16bf cur = (st & 1) ? fr1 : fr0;
        if (st + 2 < 16) {
          if (st & 1) fr1 = ldK(st + 2); else fr0 = ldK(st + 2);
        }
        int kc = st >> 2, ns = st & 3;
        sc[ns] = __builtin_amdgcn_wmma_f32_16x16x32_bf16(
            false, qa[kc], false, cur, (short)0, sc[ns], false, false);
      }
    }

    // scale + causal mask (diagonal tile only)
    const bool diag = (nt == qt);
    const int rowg = q0 + mt * 16 + hi * 8;
#pragma unroll
    for (int ns = 0; ns < 4; ++ns) {
      const int colg = n0 + ns * 16 + ln;
#pragma unroll
      for (int r = 0; r < 8; ++r) {
        float sv = sc[ns][r] * SCALING;
        if (diag && colg > rowg + r) sv = NEGINF;
        sc[ns][r] = sv;
      }
    }

    // online ghost-softmax per row
#pragma unroll
    for (int r = 0; r < 8; ++r) {
      float mx = fmaxf(fmaxf(sc[0][r], sc[1][r]), fmaxf(sc[2][r], sc[3][r]));
#pragma unroll
      for (int o = 8; o >= 1; o >>= 1) mx = fmaxf(mx, __shfl_xor(mx, o, 32));
      float mnew = fmaxf(m[r], mx);
      float corr = __expf(m[r] - mnew);
      float ps = 0.0f;
#pragma unroll
      for (int ns = 0; ns < 4; ++ns) {
        float p = __expf(sc[ns][r] - mnew);
        sc[ns][r] = p;
        ps += p;
      }
#pragma unroll
      for (int o = 8; o >= 1; o >>= 1) ps += __shfl_xor(ps, o, 32);
      l[r] = l[r] * corr + ps;
      m[r] = mnew;
#pragma unroll
      for (int ds = 0; ds < 8; ++ds) acc[ds][r] *= corr;
    }

    // stage P (bf16) into per-wave LDS for A-layout reload
    __bf16* pw = Pt + wid * (16 * PROW);
#pragma unroll
    for (int ns = 0; ns < 4; ++ns)
#pragma unroll
      for (int r = 0; r < 8; ++r)
        pw[(r + 8 * hi) * PROW + ns * 16 + ln] = (__bf16)sc[ns][r];
    asm volatile("s_wait_dscnt 0" ::: "memory");

    // ---- O += P @ V, 16 steps (c-major), 2-deep pipelined B frags ----
    const __bf16* ppb = pw + ln * PROW + fb;
    v16bf pa0 = ld_afrag(ppb,      ppb + 16);
    v16bf pa1 = ld_afrag(ppb + 32, ppb + 48);
    auto ldV = [&](int st) -> v16bf {
      int c = st >> 3, ds = st & 7;
      return ld_b16x16(Vt + (ds * 16 + ln) * VROW + c * 32 + koff);
    };
    {
      v16bf fr0 = ldV(0), fr1 = ldV(1);
#pragma unroll
      for (int st = 0; st < 16; ++st) {
        v16bf cur = (st & 1) ? fr1 : fr0;
        if (st + 2 < 16) {
          if (st & 1) fr1 = ldV(st + 2); else fr0 = ldV(st + 2);
        }
        int c = st >> 3, ds = st & 7;
        acc[ds] = __builtin_amdgcn_wmma_f32_16x16x32_bf16(
            false, (c ? pa1 : pa0), false, cur, (short)0, acc[ds], false, false);
      }
    }
    __syncthreads();
  }

  float inv[8];
#pragma unroll
  for (int r = 0; r < 8; ++r) inv[r] = 1.0f / l[r];

  if (sub == 1) {
#pragma unroll
    for (int ds = 0; ds < 8; ++ds)
#pragma unroll
      for (int r = 0; r < 8; ++r)
        Comb[(mt * 16 + r + 8 * hi) * 132 + ds * 16 + ln] = acc[ds][r] * inv[r];
  }
  __syncthreads();

  if (sub == 0) {
    const float lam = ws[0];
    float psum = 0.0f, psq = 0.0f;
    float* op = O + (((size_t)bh) * S_LEN + q0 + mt * 16 + 8 * hi) * DHEAD;
#pragma unroll
    for (int ds = 0; ds < 8; ++ds) {
#pragma unroll
      for (int r = 0; r < 8; ++r) {
        float x = acc[ds][r] * inv[r]
                - lam * Comb[(mt * 16 + r + 8 * hi) * 132 + ds * 16 + ln];
        psum += x; psq += x * x;
        op[(size_t)r * DHEAD + ds * 16 + ln] = x;
      }
    }
#pragma unroll
    for (int o = 16; o >= 1; o >>= 1) {
      psum += __shfl_xor(psum, o, 32);
      psq  += __shfl_xor(psq,  o, 32);
    }
    if (lane == 0) { atomicAdd(&s_sum, psum); atomicAdd(&s_sq, psq); }
  }
  __syncthreads();
  if (tid == 0) {
    atomicAdd(&stats[bh * 2 + 0], s_sum);
    atomicAdd(&stats[bh * 2 + 1], s_sq);
  }
}

// -------------------- kernel 3: GroupNorm + reinterpret-reshape write --------------------
__global__ __launch_bounds__(256)
void gn_write(const float* __restrict__ O, const float* __restrict__ stats,
              const float* __restrict__ gnw, const float* __restrict__ gnb,
              float* __restrict__ out) {
  const int tid  = blockIdx.x * 256 + threadIdx.x;  // B*H*D*S/4 threads
  const int t4   = tid & 511;                       // S/4
  const int rest = tid >> 9;
  const int c    = rest & 1023;                     // H*D
  const int b    = rest >> 10;
  const int t  = t4 << 2;
  const int h  = c >> 7;
  const int cl = c & 127;
  const int s  = cl * 16 + (t >> 7);
  const int d  = t & 127;
  const int bh = b * NH + h;
  const float invN = 1.0f / (float)(S_LEN * DHEAD);
  float mean = stats[bh * 2 + 0] * invN;
  float var  = stats[bh * 2 + 1] * invN - mean * mean;
  float rstd = rsqrtf(var + 1e-5f);
  float wsc = gnw[c] * rstd * 0.2f;                 // (1 - lambda_init) = 0.2
  float bsc = gnb[c] * 0.2f - mean * wsc;
  float4 x = *(const float4*)(O + (((size_t)bh) * S_LEN + s) * DHEAD + d);
  float4 y;
  y.x = x.x * wsc + bsc; y.y = x.y * wsc + bsc;
  y.z = x.z * wsc + bsc; y.w = x.w * wsc + bsc;
  *(float4*)(out + ((size_t)b * 1024 + c) * S_LEN + t) = y;
}

extern "C" void kernel_launch(void* const* d_in, const int* in_sizes, int n_in,
                              void* d_out, int out_size, void* d_ws, size_t ws_size,
                              hipStream_t stream) {
  (void)in_sizes; (void)n_in; (void)out_size; (void)ws_size;
  const float* q   = (const float*)d_in[0];
  const float* k   = (const float*)d_in[1];
  const float* v   = (const float*)d_in[2];
  const float* lq1 = (const float*)d_in[3];
  const float* lk1 = (const float*)d_in[4];
  const float* lq2 = (const float*)d_in[5];
  const float* lk2 = (const float*)d_in[6];
  const float* gnw = (const float*)d_in[7];
  const float* gnb = (const float*)d_in[8];
  float* wsf   = (float*)d_ws;
  float* stats = wsf + 8;     // 32 floats
  float* O     = wsf + 64;    // B*H*S*D = 4194304 floats

  prep_kernel<<<1, 128, 0, stream>>>(lq1, lk1, lq2, lk2, wsf);
  diff_flash<<<2 * NH * (S_LEN / BM), 256, 0, stream>>>(q, k, v, wsf, O, stats);
  gn_write<<<(2 * NH * DHEAD * S_LEN / 4) / 256, 256, 0, stream>>>(O, stats, gnw, gnb, (float*)d_out);
}
